// CustomHyperbolicLayer_55396488183984
// MI455X (gfx1250) — matrix operand order repaired and checked by reading, not compiled, not run on hardware
//
#include <hip/hip_runtime.h>

typedef __bf16 bf16_t;
typedef __attribute__((ext_vector_type(16))) __bf16 v16bf;
typedef __attribute__((ext_vector_type(8)))  __bf16 v8bf;
typedef __attribute__((ext_vector_type(4)))  __bf16 v4bf;
typedef __attribute__((ext_vector_type(8)))  float   v8f;
typedef __attribute__((ext_vector_type(4)))  int     v4i;

#define EPS_BALL 4e-3f
#define MIN_NORM 1e-15f

// ---------------------------------------------------------------------------
// CDNA5 async global->LDS copy (ASYNCcnt path), with safe fallback
// ---------------------------------------------------------------------------
#if __has_builtin(__builtin_amdgcn_global_load_async_to_lds_b128)
#define HAS_ASYNC_LDS 1
#else
#define HAS_ASYNC_LDS 0
#endif

__device__ __forceinline__ void cp16_g2l(void* lds_dst, const void* g_src) {
#if HAS_ASYNC_LDS
  typedef __attribute__((address_space(1))) v4i* g_v4i_p;  // global int4*
  typedef __attribute__((address_space(3))) v4i* l_v4i_p;  // LDS int4*
  __builtin_amdgcn_global_load_async_to_lds_b128(
      (g_v4i_p)(g_src), (l_v4i_p)(lds_dst), 0, 0);
#else
  *(uint4*)lds_dst = *(const uint4*)g_src;
#endif
}

__device__ __forceinline__ void wait_async_le4() {
#if __has_builtin(__builtin_amdgcn_s_wait_asynccnt)
  __builtin_amdgcn_s_wait_asynccnt(4);
#elif HAS_ASYNC_LDS
  asm volatile("s_wait_asynccnt 0x4" ::: "memory");
#endif
}

__device__ __forceinline__ void wait_async_0() {
#if __has_builtin(__builtin_amdgcn_s_wait_asynccnt)
  __builtin_amdgcn_s_wait_asynccnt(0);
#elif HAS_ASYNC_LDS
  asm volatile("s_wait_asynccnt 0x0" ::: "memory");
#endif
}

// ---------------------------------------------------------------------------
// reductions (wave32)
// ---------------------------------------------------------------------------
__device__ __forceinline__ float warp_reduce_sum(float v) {
#pragma unroll
  for (int off = 16; off > 0; off >>= 1) v += __shfl_xor(v, off, 32);
  return v;
}

__device__ __forceinline__ float block_reduce_sum(float v, float* sm) {
  const int lane = threadIdx.x & 31;
  const int wave = threadIdx.x >> 5;
  v = warp_reduce_sum(v);
  if (lane == 0) sm[wave] = v;
  __syncthreads();
  if (wave == 0) {
    float t = (lane < (int)(blockDim.x >> 5)) ? sm[lane] : 0.0f;
    t = warp_reduce_sum(t);
    if (lane == 0) sm[0] = t;
  }
  __syncthreads();
  float r = sm[0];
  __syncthreads();
  return r;
}

// ---------------------------------------------------------------------------
// Kernel 1: logmap0 row-scale + f32 -> bf16  (one block per row)
// ---------------------------------------------------------------------------
__global__ void __launch_bounds__(256)
prep_x_kernel(const float* __restrict__ x, bf16_t* __restrict__ A, int K) {
  __shared__ float sm[8];
  const int row = blockIdx.x;
  const float* xr = x + (size_t)row * K;
  bf16_t* ar = A + (size_t)row * K;

  float ss = 0.0f;
  for (int idx = threadIdx.x * 4; idx < K; idx += blockDim.x * 4) {
    float4 f = *(const float4*)(xr + idx);
    ss = fmaf(f.x, f.x, fmaf(f.y, f.y, fmaf(f.z, f.z, fmaf(f.w, f.w, ss))));
  }
  float n = fmaxf(sqrtf(block_reduce_sum(ss, sm)), MIN_NORM);
  float arg = fminf(n, 1.0f - 1e-7f);            // clip(sqrt_c*n, ., 1-1e-7)
  float scale = atanhf(arg) / n;                 // logmap0 row scale

  for (int idx = threadIdx.x * 4; idx < K; idx += blockDim.x * 4) {
    float4 f = *(const float4*)(xr + idx);
    v4bf o;
    o[0] = (bf16_t)(f.x * scale);
    o[1] = (bf16_t)(f.y * scale);
    o[2] = (bf16_t)(f.z * scale);
    o[3] = (bf16_t)(f.w * scale);
    *(v4bf*)(ar + idx) = o;
  }
}

// ---------------------------------------------------------------------------
// Kernel 2: W f32 -> bf16
// ---------------------------------------------------------------------------
__global__ void __launch_bounds__(256)
conv_w_kernel(const float* __restrict__ W, bf16_t* __restrict__ Wb, int n) {
  int i = (blockIdx.x * blockDim.x + threadIdx.x) * 4;
  if (i >= n) return;
  float4 f = *(const float4*)(W + i);
  v4bf o;
  o[0] = (bf16_t)f.x;
  o[1] = (bf16_t)f.y;
  o[2] = (bf16_t)f.z;
  o[3] = (bf16_t)f.w;
  *(v4bf*)(Wb + i) = o;
}

// ---------------------------------------------------------------------------
// Kernel 3: t = A @ W^T + b, v_wmma_f32_16x16x32_bf16
//   256 threads (8 waves, 2x4), block tile 128x128, K staged by 64.
//   B tile (W rows) double-buffered in LDS via async global->LDS b128
//   (DS pipe: ~256B/WMMA). A fragments read directly from global; all 4
//   N-waves issue identical A addresses -> WGP$ hits (VMEM pipe).
//   LDS row pitch padded to 72 bf16 -> conflict-free ds_load_b128.
// ---------------------------------------------------------------------------
#define TM 128
#define TN 128
#define KT 64
#define KP (KT + 8)

__global__ void __launch_bounds__(256)
gemm_kernel(const bf16_t* __restrict__ A, const bf16_t* __restrict__ B,
            const float* __restrict__ bias, float* __restrict__ out,
            int M, int N, int K) {
  __shared__ bf16_t Bs[2][TN][KP];   // 2 x 128 x 72 bf16 = 36 KB

  const int tid  = threadIdx.x;
  const int wave = tid >> 5;
  const int lane = tid & 31;
  const int half = lane >> 4;   // 0 or 1
  const int l16  = lane & 15;
  const int wm   = wave >> 2;   // 0..1 -> M offset wm*64
  const int wn   = wave & 3;    // 0..3 -> N offset wn*32
  const int blockM = blockIdx.y * TM;
  const int blockN = blockIdx.x * TN;

  const bf16_t* Bbase = B + (size_t)blockN * K;

  // A fragment row pointers (direct from global, per-lane)
  const bf16_t* arow[4];
#pragma unroll
  for (int i = 0; i < 4; ++i)
    arow[i] = A + (size_t)(blockM + wm * 64 + i * 16 + l16) * K + half * 8;

  v8f acc[4][2] = {};

  // One stage = 128 rows x 64 cols of B = 16KB.
  // 256 threads x 4 chunks x 16B -> 4 async b128 per thread per stage.
  auto stage_copy = [&](int buf, int k0) {
#pragma unroll
    for (int q = 0; q < 4; ++q) {
      const int c   = tid + 256 * q;   // 0..1023
      const int row = c >> 3;          // 0..127
      const int col = (c & 7) * 8;     // 0..56 (bf16 units)
      cp16_g2l(&Bs[buf][row][col], Bbase + (size_t)row * K + k0 + col);
    }
  };

  const int S = K / KT;
  stage_copy(0, 0);
  if (S > 1) stage_copy(1, KT);

  for (int s = 0; s < S; ++s) {
    if (s + 1 < S) wait_async_le4();   // stage s complete (s+1 still in flight)
    else           wait_async_0();
    __syncthreads();

    const int buf = s & 1;
    const int kbase = s * KT;
#pragma unroll
    for (int kk = 0; kk < KT; kk += 32) {
      v16bf af[4], bfr[2];
#pragma unroll
      for (int i = 0; i < 4; ++i) {
        const bf16_t* p = arow[i] + kbase + kk;
        v8bf lo = *(const v8bf*)p;          // K = kk + half*8 + 0..7
        v8bf hi = *(const v8bf*)(p + 16);   // K = kk + 16 + half*8 + 0..7
        af[i] = __builtin_shufflevector(lo, hi, 0, 1, 2, 3, 4, 5, 6, 7,
                                        8, 9, 10, 11, 12, 13, 14, 15);
      }
#pragma unroll
      for (int j = 0; j < 2; ++j) {
        const bf16_t* p = &Bs[buf][wn * 32 + j * 16 + l16][kk + half * 8];
        v8bf lo = *(const v8bf*)p;          // K = kk + half*8 + 0..7
        v8bf hi = *(const v8bf*)(p + 16);   // K = kk + 16 + half*8 + 0..7
        bfr[j] = __builtin_shufflevector(lo, hi, 0, 1, 2, 3, 4, 5, 6, 7,
                                         8, 9, 10, 11, 12, 13, 14, 15);
      }
#pragma unroll
      for (int i = 0; i < 4; ++i)
#pragma unroll
        for (int j = 0; j < 2; ++j)
          acc[i][j] = __builtin_amdgcn_wmma_f32_16x16x32_bf16(
              false, af[i], false, bfr[j], (short)0, acc[i][j], false, false);
    }
    __syncthreads();                    // all waves done reading buf
    if (s + 2 < S) stage_copy(buf, (s + 2) * KT);
  }

  // C/D layout: lane -> N = l16, VGPR r -> M = half*8 + r
#pragma unroll
  for (int j = 0; j < 2; ++j) {
    const int n  = blockN + wn * 32 + j * 16 + l16;
    const float bv = bias[n];
#pragma unroll
    for (int i = 0; i < 4; ++i) {
#pragma unroll
      for (int r = 0; r < 8; ++r) {
        const int m = blockM + wm * 64 + i * 16 + half * 8 + r;
        out[(size_t)m * N + n] = acc[i][j][r] + bv;
      }
    }
  }
}

// ---------------------------------------------------------------------------
// Kernel 4: fused hyperbolic epilogue, in place on d_out (one block per row)
//   y = proj(expmap0(t)); t2 = tanh(logmap0(proj(y))); out = proj(expmap0(t2))
// ---------------------------------------------------------------------------
__global__ void __launch_bounds__(256)
epilogue_kernel(float* __restrict__ out, int N) {
  __shared__ float sm[8];
  const int row = blockIdx.x;
  float* tr = out + (size_t)row * N;
  const float maxnorm = 1.0f - EPS_BALL;

  // N assumed 4096: 4 float4 per thread, fully coalesced
  float4 v[4];
  float ss = 0.0f;
#pragma unroll
  for (int it = 0; it < 4; ++it) {
    const int idx = (it * 256 + threadIdx.x) * 4;
    v[it] = *(const float4*)(tr + idx);
    float4 f = v[it];
    ss = fmaf(f.x, f.x, fmaf(f.y, f.y, fmaf(f.z, f.z, fmaf(f.w, f.w, ss))));
  }
  float nt = fmaxf(sqrtf(block_reduce_sum(ss, sm)), MIN_NORM);

  // expmap0: y = t * tanh(nt)/nt ; ||y|| = tanh(nt)
  float ny  = tanhf(nt);
  float s_a = ny / nt;
  // proj (applied twice, idempotent)
  float p = (ny > maxnorm) ? (maxnorm / ny) : 1.0f;
  float nyc = fminf(ny, maxnorm);
  // logmap0(y): atanh(clip(||y||)) / ||y||
  float arg = fminf(nyc, 1.0f - 1e-7f);
  float l = atanhf(arg) / fmaxf(nyc, MIN_NORM);
  float s_tot = s_a * p * l;

  // elementwise tanh + second norm
  float ss2 = 0.0f;
#pragma unroll
  for (int it = 0; it < 4; ++it) {
    float4 f = v[it];
    f.x = tanhf(f.x * s_tot);
    f.y = tanhf(f.y * s_tot);
    f.z = tanhf(f.z * s_tot);
    f.w = tanhf(f.w * s_tot);
    v[it] = f;
    ss2 = fmaf(f.x, f.x, fmaf(f.y, f.y, fmaf(f.z, f.z, fmaf(f.w, f.w, ss2))));
  }
  float n2 = fmaxf(sqrtf(block_reduce_sum(ss2, sm)), MIN_NORM);

  // out = proj(expmap0(t2))
  float no  = tanhf(n2);
  float s_b = no / n2;
  if (no > maxnorm) s_b *= maxnorm / no;

#pragma unroll
  for (int it = 0; it < 4; ++it) {
    const int idx = (it * 256 + threadIdx.x) * 4;
    float4 f = v[it];
    f.x *= s_b; f.y *= s_b; f.z *= s_b; f.w *= s_b;
    *(float4*)(tr + idx) = f;
  }
}

// ---------------------------------------------------------------------------
// Host launcher
// ---------------------------------------------------------------------------
extern "C" void kernel_launch(void* const* d_in, const int* in_sizes, int n_in,
                              void* d_out, int out_size, void* d_ws, size_t ws_size,
                              hipStream_t stream) {
  const float* x = (const float*)d_in[0];   // [M, K]
  const float* W = (const float*)d_in[1];   // [N, K]
  const float* b = (const float*)d_in[2];   // [N]
  float* out = (float*)d_out;               // [M, N]

  const int N = in_sizes[2];                // 4096
  const int K = in_sizes[1] / N;            // 4096
  const int M = in_sizes[0] / K;            // 8192

  bf16_t* Abf = (bf16_t*)d_ws;              // M*K bf16  (64 MB)
  bf16_t* Wbf = Abf + (size_t)M * K;        // N*K bf16  (32 MB)

  prep_x_kernel<<<M, 256, 0, stream>>>(x, Abf, K);

  const int wtot = N * K;
  conv_w_kernel<<<(wtot / 4 + 255) / 256, 256, 0, stream>>>(W, Wbf, wtot);

  dim3 grid(N / TN, M / TM);                // (32, 64)
  gemm_kernel<<<grid, 256, 0, stream>>>(Abf, Wbf, b, out, M, N, K);

  epilogue_kernel<<<M, 256, 0, stream>>>(out, N);
}
